// HHGNN_16922171147100
// MI455X (gfx1250) — matrix operand order repaired and checked by reading, not compiled, not run on hardware
//
#include <hip/hip_runtime.h>
#include <hip/hip_bf16.h>
#include <math.h>

#define NN 4096
#define TT 4
#define FF 256
#define HH 8
#define CC 16
#define PP 1024
#define DD 32
#define NT (NN * TT)

typedef __attribute__((ext_vector_type(16))) __bf16 v16bf;
typedef __attribute__((ext_vector_type(8)))  __bf16 v8bf;
typedef __attribute__((ext_vector_type(8)))  float  v8f;

// ---------------------------------------------------------------------------
// WMMA helpers (bf16 A/B, f32 accumulate; CDNA5 V_WMMA_F32_16X16X32_BF16)
// ---------------------------------------------------------------------------
__device__ __forceinline__ v8f wmma_bf16(v16bf a, v16bf b, v8f c) {
    return __builtin_amdgcn_wmma_f32_16x16x32_bf16(
        /*neg_a=*/false, a, /*neg_b=*/false, b,
        /*c_mod=*/(short)0, c, /*reuse_a=*/false, /*reuse_b=*/false);
}

// A fragment (16x32, row-major source, already offset to [row0, k0]).
// Lane m (=lane&15) holds K = base..base+7 and base+16..base+23, base=8*(lane>=16).
__device__ __forceinline__ v16bf load_a_frag(const __bf16* A, int lda, int lane) {
    const int m    = lane & 15;
    const int base = (lane < 16) ? 0 : 8;
    const __bf16* p = A + (size_t)m * lda + base;
    v16bf a;
    ((v8bf*)&a)[0] = *(const v8bf*)(p);
    ((v8bf*)&a)[1] = *(const v8bf*)(p + 16);
    return a;
}

// B fragment from a "weight-row == B-column" layout: W is (colDim x K) row-major,
// already offset to [col0, k0]. Lane c holds K = kb..kb+7 and kb+16..kb+23, kb=8*(lane>=16).
__device__ __forceinline__ v16bf load_b_frag(const __bf16* W, int ldw, int lane) {
    const int c  = lane & 15;
    const int kb = (lane < 16) ? 0 : 8;
    const __bf16* p = W + (size_t)c * ldw + kb;
    v16bf b;
    ((v8bf*)&b)[0] = *(const v8bf*)(p);
    ((v8bf*)&b)[1] = *(const v8bf*)(p + 16);
    return b;
}

// ---------------------------------------------------------------------------
// Prep kernels
// ---------------------------------------------------------------------------
__global__ void kzero(float* acc) {
    if (threadIdx.x < 3) acc[threadIdx.x] = 0.f;
}

__global__ void kcvt(const float* __restrict__ in, __bf16* __restrict__ out, int n) {
    int i = blockIdx.x * 256 + threadIdx.x;
    if (i < n) out[i] = (__bf16)in[i];
}

// featT_bf[f*N + n] = bf16(feat[n*F + f])   (makes B-columns of the big GEMM contiguous)
__global__ void ktrans(const float* __restrict__ feat, __bf16* __restrict__ featT) {
    int i = blockIdx.x * 256 + threadIdx.x;   // i = n*F + f
    int n = i >> 8, f = i & 255;
    featT[(size_t)f * NN + n] = (__bf16)feat[i];
}

// ---------------------------------------------------------------------------
// K1: feat[n,f] = sum_g feature[n,g] * theta_w[f,g] + theta_b[f]
// ---------------------------------------------------------------------------
__global__ void k1_theta(const __bf16* __restrict__ featureIn_bf,
                         const __bf16* __restrict__ theta_bf,
                         const float*  __restrict__ theta_b,
                         float* __restrict__ feat, __bf16* __restrict__ feat_bf) {
    const int n0   = blockIdx.x * 16;
    const int wave = threadIdx.x >> 5, lane = threadIdx.x & 31;
    const int c0   = wave * 32;
    v8f acc0 = {}, acc1 = {};
    for (int k0 = 0; k0 < FF; k0 += 32) {
        v16bf a  = load_a_frag(featureIn_bf + (size_t)n0 * FF + k0, FF, lane);
        v16bf b0 = load_b_frag(theta_bf + (size_t)c0 * FF + k0, FF, lane);
        v16bf b1 = load_b_frag(theta_bf + (size_t)(c0 + 16) * FF + k0, FF, lane);
        acc0 = wmma_bf16(a, b0, acc0);
        acc1 = wmma_bf16(a, b1, acc1);
    }
    const int col   = c0 + (lane & 15);
    const int rhalf = (lane < 16) ? 0 : 8;
    const float b0s = theta_b[col], b1s = theta_b[col + 16];
    for (int r = 0; r < 8; ++r) {
        int n = n0 + r + rhalf;
        float v0 = acc0[r] + b0s;
        float v1 = acc1[r] + b1s;
        feat[(size_t)n * FF + col]         = v0;
        feat[(size_t)n * FF + col + 16]    = v1;
        feat_bf[(size_t)n * FF + col]      = (__bf16)v0;
        feat_bf[(size_t)n * FF + col + 16] = (__bf16)v1;
    }
}

// ---------------------------------------------------------------------------
// K2 (dominant): ls = relu(linear*mask) generated on the fly (LDS-staged bf16),
// recon = ls @ feat (K=4096), plus per-row s1 = sum(ls), s2 = sum(ls^2),
// l1 += s1, l2 += sqrt(s2).
// ---------------------------------------------------------------------------
__global__ void k2_recon(const float* __restrict__ linear, const float* __restrict__ mask,
                         const __bf16* __restrict__ featT_bf,
                         float* __restrict__ recon, float* __restrict__ s1_arr,
                         float* __restrict__ acc3 /* [l1, l2, rerr] */) {
    __shared__ __align__(32) __bf16 Atile[16 * 32];
    const int r0   = blockIdx.x * 16;            // row index into (N*T)
    const int tid  = threadIdx.x;
    const int wave = tid >> 5, lane = tid & 31;
    const int c0   = wave * 32;
    const int tr   = tid >> 4;                   // staging: row 0..15
    const int tk   = (tid & 15) * 2;             // staging: k-pair 0..30
    float s1 = 0.f, s2 = 0.f;
    v8f acc0 = {}, acc1 = {};

    const size_t rowbase = (size_t)(r0 + tr) * NN;
    for (int k0 = 0; k0 < NN; k0 += 32) {
        size_t g = rowbase + k0 + tk;
        float l0 = linear[g], l1v = linear[g + 1];
        float m0 = mask[g],   m1v = mask[g + 1];
        __builtin_prefetch(linear + g + 256, 0, 1);   // global_prefetch (streaming)
        __builtin_prefetch(mask   + g + 256, 0, 1);
        float x0 = l0 * m0;  x0 = x0 > 0.f ? x0 : 0.f;
        float x1 = l1v * m1v; x1 = x1 > 0.f ? x1 : 0.f;
        s1 += x0 + x1;
        s2 += x0 * x0 + x1 * x1;
        Atile[tr * 32 + tk]     = (__bf16)x0;
        Atile[tr * 32 + tk + 1] = (__bf16)x1;
        __syncthreads();
        {
            const int m    = lane & 15;
            const int base = (lane < 16) ? 0 : 8;
            v16bf a;
            ((v8bf*)&a)[0] = *(const v8bf*)(Atile + m * 32 + base);
            ((v8bf*)&a)[1] = *(const v8bf*)(Atile + m * 32 + base + 16);
            v16bf b0 = load_b_frag(featT_bf + (size_t)c0 * NN + k0, NN, lane);
            v16bf b1 = load_b_frag(featT_bf + (size_t)(c0 + 16) * NN + k0, NN, lane);
            acc0 = wmma_bf16(a, b0, acc0);
            acc1 = wmma_bf16(a, b1, acc1);
        }
        __syncthreads();
    }
    // store recon stripe
    const int col   = c0 + (lane & 15);
    const int rhalf = (lane < 16) ? 0 : 8;
    for (int r = 0; r < 8; ++r) {
        size_t row = (size_t)(r0 + r + rhalf);
        recon[row * FF + col]      = acc0[r];
        recon[row * FF + col + 16] = acc1[r];
    }
    // reduce s1/s2 across the 16 threads sharing a staging row (contiguous half-wave)
    for (int off = 1; off < 16; off <<= 1) {
        s1 += __shfl_xor(s1, off);
        s2 += __shfl_xor(s2, off);
    }
    if ((tid & 15) == 0) {
        s1_arr[r0 + tr] = s1;
        atomicAdd(&acc3[0], s1);           // l1
        atomicAdd(&acc3[1], sqrtf(s2));    // l2
    }
}

// ---------------------------------------------------------------------------
// K3: for (s,j): out = feat @ rp[s,j]^T ; rerr += sqrt(sum_f (out-recon)^2)
// for rows with nmm[n]==s. inp_proj never materialized.
// ---------------------------------------------------------------------------
__global__ void k3_reconerr(const __bf16* __restrict__ feat_bf,
                            const __bf16* __restrict__ rp_bf,
                            const float* __restrict__ recon,
                            const int* __restrict__ nmm,
                            float* __restrict__ acc3) {
    __shared__ float rowsq[16];
    const int sj = blockIdx.y;               // s*T + j
    const int s = sj >> 2, j = sj & 3;
    const int n0 = blockIdx.x * 16;
    const __bf16* W = rp_bf + (size_t)sj * FF * FF;
    if (threadIdx.x < 16) rowsq[threadIdx.x] = 0.f;
    __syncthreads();
    const int wave = threadIdx.x >> 5, lane = threadIdx.x & 31;
    const int c0 = wave * 32;
    v8f acc0 = {}, acc1 = {};
    for (int k0 = 0; k0 < FF; k0 += 32) {
        v16bf a  = load_a_frag(feat_bf + (size_t)n0 * FF + k0, FF, lane);
        v16bf b0 = load_b_frag(W + (size_t)c0 * FF + k0, FF, lane);
        v16bf b1 = load_b_frag(W + (size_t)(c0 + 16) * FF + k0, FF, lane);
        acc0 = wmma_bf16(a, b0, acc0);
        acc1 = wmma_bf16(a, b1, acc1);
    }
    const int col   = c0 + (lane & 15);
    const int rhalf = (lane < 16) ? 0 : 8;
    for (int r = 0; r < 8; ++r) {
        int lr = r + rhalf;
        size_t base = ((size_t)(n0 + lr) * TT + j) * FF;
        float d0 = acc0[r] - recon[base + col];
        float d1 = acc1[r] - recon[base + col + 16];
        atomicAdd(&rowsq[lr], d0 * d0 + d1 * d1);   // ds_add_f32
    }
    __syncthreads();
    if (threadIdx.x < 16) {
        int n = n0 + threadIdx.x;
        if (nmm[n] == s) atomicAdd(&acc3[2], sqrtf(rowsq[threadIdx.x]));
    }
}

// ---------------------------------------------------------------------------
// K4: edge_fea[n,t,f] = (recon + feat[n,f]) / (s1[n*T+t] + 1)   (ff == feat)
// ---------------------------------------------------------------------------
__global__ void k4_edgefea(const float* __restrict__ recon, const float* __restrict__ feat,
                           const float* __restrict__ s1_arr,
                           float* __restrict__ edge_fea, __bf16* __restrict__ edge_fea_bf) {
    size_t i = (size_t)blockIdx.x * 256 + threadIdx.x;   // over N*T*F
    int f = (int)(i & 255);
    int row = (int)(i >> 8);
    int n = row >> 2;
    float v = (recon[i] + feat[(size_t)n * FF + f]) / (s1_arr[row] + 1.0f);
    edge_fea[i]    = v;
    edge_fea_bf[i] = (__bf16)v;
}

// ---------------------------------------------------------------------------
// K5: node_projd[n,f] = sum_g mh_node_proj[nmm[n],f,g]*feat[n,g]
// ---------------------------------------------------------------------------
__global__ void k5_nodeproj(const __bf16* __restrict__ feat_bf,
                            const __bf16* __restrict__ np_bf,
                            const int* __restrict__ nmm,
                            float* __restrict__ node_projd) {
    const int s  = blockIdx.y;
    const int n0 = blockIdx.x * 16;
    const __bf16* W = np_bf + (size_t)s * FF * FF;
    const int wave = threadIdx.x >> 5, lane = threadIdx.x & 31;
    const int c0 = wave * 32;
    v8f acc0 = {}, acc1 = {};
    for (int k0 = 0; k0 < FF; k0 += 32) {
        v16bf a  = load_a_frag(feat_bf + (size_t)n0 * FF + k0, FF, lane);
        v16bf b0 = load_b_frag(W + (size_t)c0 * FF + k0, FF, lane);
        v16bf b1 = load_b_frag(W + (size_t)(c0 + 16) * FF + k0, FF, lane);
        acc0 = wmma_bf16(a, b0, acc0);
        acc1 = wmma_bf16(a, b1, acc1);
    }
    const int col   = c0 + (lane & 15);
    const int rhalf = (lane < 16) ? 0 : 8;
    for (int r = 0; r < 8; ++r) {
        int n = n0 + r + rhalf;
        if (nmm[n] == s) {
            node_projd[(size_t)n * FF + col]      = acc0[r];
            node_projd[(size_t)n * FF + col + 16] = acc1[r];
        }
    }
}

// ---------------------------------------------------------------------------
// K6: edge_projd[n,j,f] = sum_g mh_edge_proj[nmm[n],j,f,g]*edge_fea[n,j,g]
// ---------------------------------------------------------------------------
__global__ void k6_edgeproj(const __bf16* __restrict__ edge_fea_bf,
                            const __bf16* __restrict__ ep_bf,
                            const int* __restrict__ nmm,
                            float* __restrict__ edge_projd) {
    const int sj = blockIdx.y;
    const int s = sj >> 2, j = sj & 3;
    const int n0 = blockIdx.x * 16;
    const __bf16* A = edge_fea_bf + (size_t)j * FF;     // row n at n*(T*F)
    const __bf16* W = ep_bf + (size_t)sj * FF * FF;
    const int wave = threadIdx.x >> 5, lane = threadIdx.x & 31;
    const int c0 = wave * 32;
    v8f acc0 = {}, acc1 = {};
    for (int k0 = 0; k0 < FF; k0 += 32) {
        v16bf a  = load_a_frag(A + (size_t)n0 * (TT * FF) + k0, TT * FF, lane);
        v16bf b0 = load_b_frag(W + (size_t)c0 * FF + k0, FF, lane);
        v16bf b1 = load_b_frag(W + (size_t)(c0 + 16) * FF + k0, FF, lane);
        acc0 = wmma_bf16(a, b0, acc0);
        acc1 = wmma_bf16(a, b1, acc1);
    }
    const int col   = c0 + (lane & 15);
    const int rhalf = (lane < 16) ? 0 : 8;
    for (int r = 0; r < 8; ++r) {
        int n = n0 + r + rhalf;
        if (nmm[n] == s) {
            size_t base = ((size_t)n * TT + j) * FF;
            edge_projd[base + col]      = acc0[r];
            edge_projd[base + col + 16] = acc1[r];
        }
    }
}

// ---------------------------------------------------------------------------
// K7: per-node attention (softmax over H) + relu + sum over T -> node_rep
// One wave per node; 8 nodes per 256-thread block.
// ---------------------------------------------------------------------------
__global__ void k7_attn(const float* __restrict__ node_projd,
                        const float* __restrict__ edge_projd,
                        const float* __restrict__ edge_fea,
                        float* __restrict__ node_rep) {
    __shared__ float wbuf[8][32];
    const int wave = threadIdx.x >> 5, lane = threadIdx.x & 31;
    const int n = blockIdx.x * 8 + wave;
    const int h = lane >> 2, t = lane & 3;
    const float* npp = node_projd + (size_t)n * FF + h * DD;
    const float* epr = edge_projd + ((size_t)n * TT + t) * FF + h * DD;
    float sc = 0.f;
    for (int d = 0; d < DD; ++d) sc += npp[d] * epr[d];
    sc *= 0.1767766952966369f;   // 1/sqrt(D)
    // softmax over H (lanes differing in bits 2..4)
    float m = sc;
    m = fmaxf(m, __shfl_xor(m, 4));
    m = fmaxf(m, __shfl_xor(m, 8));
    m = fmaxf(m, __shfl_xor(m, 16));
    float e = __expf(sc - m);
    float ss = e;
    ss += __shfl_xor(ss, 4);
    ss += __shfl_xor(ss, 8);
    ss += __shfl_xor(ss, 16);
    wbuf[wave][lane] = e / ss;   // index h*4+t == lane
    __syncthreads();
    for (int fi = 0; fi < 8; ++fi) {
        int f = lane + fi * 32;
        int hh = f >> 5;
        float acc = 0.f;
        for (int tt = 0; tt < TT; ++tt) {
            float w = wbuf[wave][hh * 4 + tt];
            float v = w * edge_fea[((size_t)n * TT + tt) * FF + f];
            acc += fmaxf(v, 0.f);
        }
        node_rep[(size_t)n * FF + f] = acc;
    }
}

// ---------------------------------------------------------------------------
// K8: logits -> sigmoid -> softmax (over C=16) + assemble recon_loss scalar
// ---------------------------------------------------------------------------
__global__ void k8_pred(const float* __restrict__ node_rep,
                        const int* __restrict__ node_idx,
                        const float* __restrict__ pred_w,
                        const float* __restrict__ pred_b,
                        const float* __restrict__ acc3,
                        float* __restrict__ out) {
    if (blockIdx.x == 0 && threadIdx.x == 0)
        out[PP * CC] = 0.1f * (0.01f * acc3[2] + 0.2f * acc3[1] + acc3[0]);
    const int p = blockIdx.x * 16 + (threadIdx.x >> 4);
    const int c = threadIdx.x & 15;
    const int n = node_idx[p];
    float acc = pred_b[c];
    const float* nr = node_rep + (size_t)n * FF;
    const float* pw = pred_w + (size_t)c * FF;
    for (int f = 0; f < FF; ++f) acc += nr[f] * pw[f];
    float sg = 1.f / (1.f + __expf(-acc));
    float m = sg;
    for (int off = 1; off < 16; off <<= 1) m = fmaxf(m, __shfl_xor(m, off));
    float e = __expf(sg - m);
    float s = e;
    for (int off = 1; off < 16; off <<= 1) s += __shfl_xor(s, off);
    out[(size_t)p * CC + c] = e / s;
}

// ---------------------------------------------------------------------------
// Host-side orchestration
// ---------------------------------------------------------------------------
extern "C" void kernel_launch(void* const* d_in, const int* in_sizes, int n_in,
                              void* d_out, int out_size, void* d_ws, size_t ws_size,
                              hipStream_t stream) {
    const float* feature      = (const float*)d_in[0];
    const float* mask         = (const float*)d_in[1];
    const int*   nmm          = (const int*)d_in[2];
    const int*   node_idx     = (const int*)d_in[3];
    const float* theta_w      = (const float*)d_in[4];
    const float* theta_b      = (const float*)d_in[5];
    const float* linear       = (const float*)d_in[6];
    const float* recon_proj   = (const float*)d_in[7];
    const float* mh_node_proj = (const float*)d_in[8];
    const float* mh_edge_proj = (const float*)d_in[9];
    const float* pred_w       = (const float*)d_in[10];
    const float* pred_b       = (const float*)d_in[11];
    float* out = (float*)d_out;

    char* ws = (char*)d_ws;
    size_t off = 0;
    auto alloc = [&](size_t bytes) -> char* {
        char* p = ws + off;
        off = (off + bytes + 255) & ~(size_t)255;
        return p;
    };
    float*  feat         = (float*)alloc((size_t)NN * FF * 4);
    float*  recon        = (float*)alloc((size_t)NT * FF * 4);
    float*  edge_fea     = (float*)alloc((size_t)NT * FF * 4);
    float*  edge_projd   = (float*)alloc((size_t)NT * FF * 4);
    float*  node_projd   = (float*)alloc((size_t)NN * FF * 4);
    float*  s1_arr       = (float*)alloc((size_t)NT * 4);
    float*  acc3         = (float*)alloc(256);
    __bf16* feat_bf      = (__bf16*)alloc((size_t)NN * FF * 2);
    __bf16* featT_bf     = (__bf16*)alloc((size_t)NN * FF * 2);
    __bf16* featureIn_bf = (__bf16*)alloc((size_t)NN * FF * 2);
    __bf16* theta_bf     = (__bf16*)alloc((size_t)FF * FF * 2);
    __bf16* rp_bf        = (__bf16*)alloc((size_t)16 * FF * FF * 2);
    __bf16* np_bf        = (__bf16*)alloc((size_t)TT * FF * FF * 2);
    __bf16* ep_bf        = (__bf16*)alloc((size_t)16 * FF * FF * 2);
    __bf16* edge_fea_bf  = (__bf16*)alloc((size_t)NT * FF * 2);

    float* node_rep = out + PP * CC + 1;   // output region for node_rep

    kzero<<<1, 32, 0, stream>>>(acc3);
    kcvt<<<(NN * FF + 255) / 256, 256, 0, stream>>>(feature, featureIn_bf, NN * FF);
    kcvt<<<(FF * FF + 255) / 256, 256, 0, stream>>>(theta_w, theta_bf, FF * FF);
    kcvt<<<(16 * FF * FF + 255) / 256, 256, 0, stream>>>(recon_proj, rp_bf, 16 * FF * FF);
    kcvt<<<(TT * FF * FF + 255) / 256, 256, 0, stream>>>(mh_node_proj, np_bf, TT * FF * FF);
    kcvt<<<(16 * FF * FF + 255) / 256, 256, 0, stream>>>(mh_edge_proj, ep_bf, 16 * FF * FF);

    k1_theta<<<NN / 16, 256, 0, stream>>>(featureIn_bf, theta_bf, theta_b, feat, feat_bf);
    ktrans<<<(NN * FF) / 256, 256, 0, stream>>>(feat, featT_bf);
    k2_recon<<<NT / 16, 256, 0, stream>>>(linear, mask, featT_bf, recon, s1_arr, acc3);
    k3_reconerr<<<dim3(NN / 16, 16), 256, 0, stream>>>(feat_bf, rp_bf, recon, nmm, acc3);
    k4_edgefea<<<(NT * FF) / 256, 256, 0, stream>>>(recon, feat, s1_arr, edge_fea, edge_fea_bf);
    k5_nodeproj<<<dim3(NN / 16, TT), 256, 0, stream>>>(feat_bf, np_bf, nmm, node_projd);
    k6_edgeproj<<<dim3(NN / 16, 16), 256, 0, stream>>>(edge_fea_bf, ep_bf, nmm, edge_projd);
    k7_attn<<<NN / 8, 256, 0, stream>>>(node_projd, edge_projd, edge_fea, node_rep);
    k8_pred<<<PP / 16, 256, 0, stream>>>(node_rep, node_idx, pred_w, pred_b, acc3, out);

    (void)in_sizes; (void)n_in; (void)out_size; (void)ws_size;
}
